// MIL_Attention_FC_surv_69106023793083
// MI455X (gfx1250) — matrix operand order repaired and checked by reading, not compiled, not run on hardware
//
#include <hip/hip_runtime.h>
#include <math.h>

// ---------------------------------------------------------------------------
// MIL attention GCN survival head for gfx1250 (MI455X).
// - All large GEMMs: v_wmma_f32_16x16x32_bf16 (bf16 in, f32 accumulate)
// - Tile staging: Tensor Data Mover (tensor_load_to_lds), DOUBLE-BUFFERED so
//   the DMA overlaps WMMA; drained with s_wait_tensorcnt after the compute
//   barrier. Manual-copy fallback if the builtin is absent.
// - Every GEMM consumes B transposed (rows contiguous along K) so both tiles
//   are plain 2D strided DMAs.
// ---------------------------------------------------------------------------

typedef __attribute__((ext_vector_type(8)))  __bf16 bf16x8;
typedef __attribute__((ext_vector_type(16))) __bf16 bf16x16;
typedef __attribute__((ext_vector_type(8)))  float  f32x8;
typedef __attribute__((ext_vector_type(4)))  unsigned int u32x4;
typedef __attribute__((ext_vector_type(4)))  int i32x4;
typedef __attribute__((ext_vector_type(8)))  int i32x8;
typedef __bf16 bf16;

#define NP 6144
#define LF 1024
#define DH 512
#define ALPHA_C 0.4f
#define THRESH_C 0.5f

#define BM 128
#define BN 128
#define BK 32
#define BKP 40   // padded LDS K-stride (shorts): 64B row + 16B pad, 16B aligned
#define TILE_BYTES (BM * BKP * 2)

#if __has_builtin(__builtin_amdgcn_tensor_load_to_lds)
#define USE_TDM 1
#else
#define USE_TDM 0
#endif

// ------------------------- block reductions --------------------------------
template <int MODE> // 0 sum, 1 max, 2 min
__device__ __forceinline__ float block_reduce(float v, float* sb) {
  const int t = threadIdx.x;
#pragma unroll
  for (int o = 16; o > 0; o >>= 1) {
    float u = __shfl_down(v, o, 32);
    if (MODE == 0) v += u; else if (MODE == 1) v = fmaxf(v, u); else v = fminf(v, u);
  }
  if ((t & 31) == 0) sb[t >> 5] = v;
  __syncthreads();
  const int nw = blockDim.x >> 5;
  if (t < 32) {
    v = (t < nw) ? sb[t]
                 : (MODE == 0 ? 0.0f : (MODE == 1 ? -3.402823466e38f : 3.402823466e38f));
#pragma unroll
    for (int o = 16; o > 0; o >>= 1) {
      float u = __shfl_down(v, o, 32);
      if (MODE == 0) v += u; else if (MODE == 1) v = fmaxf(v, u); else v = fminf(v, u);
    }
    if (t == 0) sb[0] = v;
  }
  __syncthreads();
  float r = sb[0];
  __syncthreads();
  return r;
}

__device__ __forceinline__ float sigmoidf(float x) { return 1.0f / (1.0f + expf(-x)); }

// ------------------------- TDM tile load -----------------------------------
// Load a [rows=128] x [32 bf16] tile, global row stride = strideElems (bf16),
// into LDS at byte offset lds_off with the BKP padded layout
// (pad_interval = 16 DWORDs -> pad after every 64B row, pad_amount = 4 DWORDs).
#if USE_TDM
__device__ __forceinline__ void tdm_load_tile(unsigned lds_off, const bf16* gsrc,
                                              int strideElems) {
  unsigned long long ga = (unsigned long long)gsrc;
  u32x4 g0;
  g0[0] = 1u;                                         // count=1, user D#
  g0[1] = lds_off;                                    // lds_addr (bytes)
  g0[2] = (unsigned)(ga & 0xFFFFFFFFu);               // global_addr lo
  g0[3] = (unsigned)((ga >> 32) & 0x1FFFFFFu) | (2u << 30); // addr hi | type=2
  i32x8 g1;
  // data_size=1 (2B), pad_enable=1, pad_interval=3 (16 DW), pad_amount=3 (4 DW)
  g1[0] = (int)((1u << 16) | (1u << 20) | (3u << 22) | (3u << 25));
  const unsigned td0 = 0x7FFFFFFFu, td1 = 0x7FFFFFFFu;  // huge tensor dims (no OOB)
  g1[1] = (int)((td0 & 0xFFFFu) << 16);               // tensor_dim0 lo16 (<<16)
  g1[2] = (int)((td0 >> 16) | ((td1 & 0xFFFFu) << 16));
  g1[3] = (int)((td1 >> 16) | (32u << 16));           // tile_dim0 = 32 elems
  g1[4] = 128;                                        // tile_dim1 = 128, tile_dim2 = 0
  g1[5] = strideElems;                                // tensor_dim0_stride lo32
  g1[6] = 0;                                          // stride hi, dim1_stride lo
  g1[7] = 0;
  i32x4 gz;  gz[0] = gz[1] = gz[2] = gz[3] = 0;       // 2D tensor: groups 2/3 unused
#if defined(__clang_major__) && (__clang_major__ >= 23)
  i32x8 gz8; for (int i = 0; i < 8; i++) gz8[i] = 0;
  __builtin_amdgcn_tensor_load_to_lds(g0, g1, gz, gz, gz8, 0);
#else
  __builtin_amdgcn_tensor_load_to_lds(g0, g1, gz, gz, 0);
#endif
}
__device__ __forceinline__ void tdm_wait0() {
#if __has_builtin(__builtin_amdgcn_s_wait_tensorcnt)
  __builtin_amdgcn_s_wait_tensorcnt((short)0);
#else
  asm volatile("s_wait_tensorcnt 0x0" ::: "memory");
#endif
}
#endif

// ---- fragment gather (per ISA 16-bit layouts) + 8 WMMAs on one tile -------
__device__ __forceinline__ void compute_tile(const bf16* __restrict__ tA,
                                             const bf16* __restrict__ tB,
                                             int wm, int wn, int lmod, int lhalf,
                                             f32x8 acc[4][2]) {
  bf16x16 afr[4];
#pragma unroll
  for (int mi = 0; mi < 4; mi++) {
    int r = wm * 64 + mi * 16 + lmod;
    const bf16* p = &tA[r * BKP + lhalf * 8];
    bf16x8 lo = *(const bf16x8*)p;          // K = khalf .. khalf+7
    bf16x8 hi = *(const bf16x8*)(p + 16);   // K = khalf+16 .. khalf+23
    afr[mi] = __builtin_shufflevector(lo, hi, 0,1,2,3,4,5,6,7,8,9,10,11,12,13,14,15);
  }
  bf16x16 bfr[2];
#pragma unroll
  for (int ni = 0; ni < 2; ni++) {
    int c = wn * 32 + ni * 16 + lmod;
    const bf16* p = &tB[c * BKP + lhalf * 8];
    bf16x8 lo = *(const bf16x8*)p;          // K = khalf .. khalf+7
    bf16x8 hi = *(const bf16x8*)(p + 16);   // K = khalf+16 .. khalf+23
    bfr[ni] = __builtin_shufflevector(lo, hi, 0,1,2,3,4,5,6,7,8,9,10,11,12,13,14,15);
  }
#pragma unroll
  for (int mi = 0; mi < 4; mi++)
#pragma unroll
    for (int ni = 0; ni < 2; ni++)
      acc[mi][ni] = __builtin_amdgcn_wmma_f32_16x16x32_bf16(
          false, afr[mi], false, bfr[ni], (short)0, acc[mi][ni], false, false);
}

// ------------------------- bf16 WMMA GEMM ----------------------------------
// C[M,Nn] (f32, += bias[col]) = A[M,K](bf16) @ B, with B supplied TRANSPOSED:
// Bt is [Nn,K] row-major (row n of Bt = column n of B, contiguous along K).
__global__ __launch_bounds__(256)
void gemm_bf16_kernel(const bf16* __restrict__ A, const bf16* __restrict__ Bt,
                      float* __restrict__ C, const float* __restrict__ bias,
                      int M, int Nn, int K) {
  __shared__ alignas(16) bf16 sA[2][BM * BKP];
  __shared__ alignas(16) bf16 sB[2][BN * BKP];   // sB[buf][n*BKP + k]

  const int t = threadIdx.x;
  const int lane = t & 31;
  const int wave = t >> 5;
  const int wm = wave >> 2;      // 0..1  (64-row slab)
  const int wn = wave & 3;       // 0..3  (32-col slab)
  const int lmod = lane & 15;
  const int lhalf = lane >> 4;
  const int bm = blockIdx.y * BM;
  const int bn = blockIdx.x * BN;

  f32x8 acc[4][2] = {};

#if USE_TDM
  const unsigned ldsA0 = (unsigned)(unsigned long long)&sA[0][0];
  const unsigned ldsB0 = (unsigned)(unsigned long long)&sB[0][0];

  // prologue: DMA tile 0 into buffer 0
  if (wave == 0) {
    tdm_load_tile(ldsA0, A + (size_t)bm * K, K);
    tdm_load_tile(ldsB0, Bt + (size_t)bn * K, K);
    tdm_wait0();
  }
  __syncthreads();

  int cur = 0;
  for (int kb = 0; kb < K; kb += BK) {
    // issue DMA for tile kb+BK into the other buffer (prev barrier freed it)
    if (wave == 0 && (kb + BK < K)) {
      int nxt = cur ^ 1;
      tdm_load_tile(ldsA0 + (unsigned)nxt * TILE_BYTES, A + (size_t)bm * K + kb + BK, K);
      tdm_load_tile(ldsB0 + (unsigned)nxt * TILE_BYTES, Bt + (size_t)bn * K + kb + BK, K);
    }
    // compute on current buffer while the DMA streams the next tile
    compute_tile(&sA[cur][0], &sB[cur][0], wm, wn, lmod, lhalf, acc);
    __syncthreads();               // everyone done reading buffer `cur`
    if (wave == 0) tdm_wait0();    // next tile fully landed in LDS
    __syncthreads();               // publish to all waves
    cur ^= 1;
  }
#else
  const int ra = t >> 1;         // 0..127 tile row for copies
  const int ha = t & 1;          // 0/1 half of the 32-wide K slice
  for (int kb = 0; kb < K; kb += BK) {
    {
      const bf16* src = A + (size_t)(bm + ra) * K + kb + ha * 16;
      uint4 d0 = ((const uint4*)src)[0];
      uint4 d1 = ((const uint4*)src)[1];
      uint4* dst = (uint4*)&sA[0][ra * BKP + ha * 16];
      dst[0] = d0; dst[1] = d1;
    }
    {
      const bf16* src = Bt + (size_t)(bn + ra) * K + kb + ha * 16;
      uint4 d0 = ((const uint4*)src)[0];
      uint4 d1 = ((const uint4*)src)[1];
      uint4* dst = (uint4*)&sB[0][ra * BKP + ha * 16];
      dst[0] = d0; dst[1] = d1;
    }
    __syncthreads();
    compute_tile(&sA[0][0], &sB[0][0], wm, wn, lmod, lhalf, acc);
    __syncthreads();
  }
#endif

  // ---- epilogue: C layout (VGPR e -> M = e + lhalf*8, lane%16 -> N) ----
#pragma unroll
  for (int mi = 0; mi < 4; mi++) {
    int row0 = bm + wm * 64 + mi * 16 + lhalf * 8;
#pragma unroll
    for (int ni = 0; ni < 2; ni++) {
      int col = bn + wn * 32 + ni * 16 + lmod;
      float bv = bias ? bias[col] : 0.0f;
#pragma unroll
      for (int e = 0; e < 8; e++)
        C[(size_t)(row0 + e) * Nn + col] = acc[mi][ni][e] + bv;
    }
  }
}

// ------------------------- small kernels -----------------------------------
__global__ void cast_f32_bf16(const float* __restrict__ in, bf16* __restrict__ out,
                              long long n) {
  for (long long i = (long long)blockIdx.x * blockDim.x + threadIdx.x; i < n;
       i += (long long)gridDim.x * blockDim.x)
    out[i] = (bf16)in[i];
}

// out[c*R + r] = (bf16) in[r*C + c]   (in: [R,C] f32 -> out: [C,R] bf16)
__global__ __launch_bounds__(256)
void transpose_cast(const float* __restrict__ in, bf16* __restrict__ out,
                    int R, int C) {
  long long n = (long long)R * C;
  for (long long i = (long long)blockIdx.x * 256 + threadIdx.x; i < n;
       i += (long long)gridDim.x * 256) {
    int r = (int)(i / C), c = (int)(i % C);
    out[(size_t)c * R + r] = (bf16)in[i];
  }
}

__global__ __launch_bounds__(256)
void rownorm_l2(const float* __restrict__ x, bf16* __restrict__ out) {
  __shared__ float sb[32];
  const int i = blockIdx.x;
  const float* row = x + (size_t)i * LF;
  float ss = 0.0f;
  for (int c = threadIdx.x; c < LF; c += 256) { float v = row[c]; ss += v * v; }
  ss = block_reduce<0>(ss, sb);
  float inv = 1.0f / fmaxf(sqrtf(ss), 1e-12f);
  for (int c = threadIdx.x; c < LF; c += 256)
    out[(size_t)i * LF + c] = (bf16)(row[c] * inv);
}

__global__ __launch_bounds__(256)
void minmax_partial(const float* __restrict__ x, size_t n,
                    float* __restrict__ pmin, float* __restrict__ pmax) {
  __shared__ float sb[32];
  float mn = 3.402823466e38f, mx = -3.402823466e38f;
  for (size_t i = (size_t)blockIdx.x * 256 + threadIdx.x; i < n;
       i += (size_t)gridDim.x * 256) {
    float v = x[i]; mn = fminf(mn, v); mx = fmaxf(mx, v);
  }
  mn = block_reduce<2>(mn, sb);
  mx = block_reduce<1>(mx, sb);
  if (threadIdx.x == 0) { pmin[blockIdx.x] = mn; pmax[blockIdx.x] = mx; }
}

__global__ __launch_bounds__(256)
void minmax_final(const float* __restrict__ pmin, const float* __restrict__ pmax,
                  int n, float* __restrict__ out2) {
  __shared__ float sb[32];
  float mn = 3.402823466e38f, mx = -3.402823466e38f;
  for (int i = threadIdx.x; i < n; i += 256) {
    mn = fminf(mn, pmin[i]); mx = fmaxf(mx, pmax[i]);
  }
  mn = block_reduce<2>(mn, sb);
  mx = block_reduce<1>(mx, sb);
  if (threadIdx.x == 0) { out2[0] = mn; out2[1] = mx; }
}

// agg_i = (tanh(apre) * sigmoid(bpre)) . wc + bc ;  s_i = sigmoid(agg_i)
__global__ __launch_bounds__(256)
void agg_kernel(const float* __restrict__ apre, const float* __restrict__ bpre,
                const float* __restrict__ wc, const float* __restrict__ bc,
                float* __restrict__ agg, float* __restrict__ s) {
  __shared__ float sb[32];
  const int i = blockIdx.x, c = threadIdx.x;
  float av = tanhf(apre[(size_t)i * 256 + c]);
  float bv = sigmoidf(bpre[(size_t)i * 256 + c]);
  float v = block_reduce<0>(av * bv * wc[c], sb);
  if (c == 0) { float g = v + bc[0]; agg[i] = g; s[i] = sigmoidf(g); }
}

__global__ __launch_bounds__(256)
void vec_stats(const float* __restrict__ s, int n, float* __restrict__ out3) {
  __shared__ float sb[32];
  float mn = 3.402823466e38f, mx = -3.402823466e38f, sm = 0.0f;
  for (int i = threadIdx.x; i < n; i += 256) {
    float v = s[i]; mn = fminf(mn, v); mx = fmaxf(mx, v); sm += v;
  }
  mn = block_reduce<2>(mn, sb);
  mx = block_reduce<1>(mx, sb);
  sm = block_reduce<0>(sm, sb);
  if (threadIdx.x == 0) { out3[0] = mn; out3[1] = mx; out3[2] = sm; }
}

// aw row: a_ij = c1 + c2*s_i*s_j + maps_ij ; L1-normalize row; write bf16
// (min/max of s*s^T derived analytically: all s>0 -> min=(min s)^2, max=(max s)^2)
__global__ __launch_bounds__(256)
void aw_build(const float* __restrict__ maps, const float* __restrict__ s,
              const float* __restrict__ st3, bf16* __restrict__ awbf) {
  __shared__ float sb[32];
  const int i = blockIdx.x;
  float smin = st3[0], smax = st3[1];
  float mn2 = smin * smin, mx2 = smax * smax;
  float c2 = (1.0f - ALPHA_C) / fmaxf(mx2 - mn2, 1e-20f);
  float c1 = ALPHA_C - c2 * mn2;
  float si = s[i];
  const float* mrow = maps + (size_t)i * NP;
  float sa = 0.0f;
  for (int j = threadIdx.x; j < NP; j += 256)
    sa += fabsf(c1 + c2 * si * s[j] + mrow[j]);
  sa = block_reduce<0>(sa, sb);
  float inv = 1.0f / fmaxf(sa, 1e-12f);
  for (int j = threadIdx.x; j < NP; j += 256)
    awbf[(size_t)i * NP + j] = (bf16)((c1 + c2 * si * s[j] + mrow[j]) * inv);
}

// threshold cs, L1-normalize (uniform 1/cnt per row), fuse trans = cs_l1@agg / sqrt(N)
__global__ __launch_bounds__(256)
void cs_thresh(const float* __restrict__ cs, const float* __restrict__ mm2,
               const float* __restrict__ agg, bf16* __restrict__ csbf,
               float* __restrict__ trans) {
  __shared__ float sb[32];
  const int i = blockIdx.x;
  float mn = mm2[0], mx = mm2[1];
  float thr = mn + THRESH_C * (mx - mn);
  const float* row = cs + (size_t)i * NP;
  float cnt = 0.0f, dot = 0.0f;
  for (int j = threadIdx.x; j < NP; j += 256) {
    if (row[j] >= thr) { cnt += 1.0f; dot += agg[j]; }
  }
  cnt = block_reduce<0>(cnt, sb);
  dot = block_reduce<0>(dot, sb);
  float inv = 1.0f / fmaxf(cnt, 1e-12f);
  for (int j = threadIdx.x; j < NP; j += 256)
    csbf[(size_t)i * NP + j] = (bf16)((row[j] >= thr) ? inv : 0.0f);
  if (threadIdx.x == 0) trans[i] = inv * dot * rsqrtf((float)NP);
}

// LayerNorm(512) + ReLU, dual output (f32 + bf16)
__global__ __launch_bounds__(256)
void lnrelu(const float* __restrict__ in, const float* __restrict__ g,
            const float* __restrict__ b, float* __restrict__ outf,
            bf16* __restrict__ outb) {
  __shared__ float sb[32];
  const int i = blockIdx.x, c0 = threadIdx.x, c1 = threadIdx.x + 256;
  const float* row = in + (size_t)i * DH;
  float v0 = row[c0], v1 = row[c1];
  float s1 = block_reduce<0>(v0 + v1, sb);
  float s2 = block_reduce<0>(v0 * v0 + v1 * v1, sb);
  float m = s1 * (1.0f / DH);
  float var = s2 * (1.0f / DH) - m * m;
  float inv = rsqrtf(var + 1e-5f);
  float y0 = fmaxf((v0 - m) * inv * g[c0] + b[c0], 0.0f);
  float y1 = fmaxf((v1 - m) * inv * g[c1] + b[c1], 0.0f);
  outf[(size_t)i * DH + c0] = y0; outf[(size_t)i * DH + c1] = y1;
  outb[(size_t)i * DH + c0] = (bf16)y0; outb[(size_t)i * DH + c1] = (bf16)y1;
}

__global__ __launch_bounds__(1024)
void softmax_fw(const float* __restrict__ trans, const float* __restrict__ agg,
                float* __restrict__ fw) {
  __shared__ float sb[32];
  float mx = -3.402823466e38f;
  for (int i = threadIdx.x; i < NP; i += 1024)
    mx = fmaxf(mx, 0.3f * trans[i] + 0.7f * agg[i]);
  mx = block_reduce<1>(mx, sb);
  float se = 0.0f;
  for (int i = threadIdx.x; i < NP; i += 1024)
    se += expf(0.3f * trans[i] + 0.7f * agg[i] - mx);
  se = block_reduce<0>(se, sb);
  float inv = 1.0f / se;
  for (int i = threadIdx.x; i < NP; i += 1024)
    fw[i] = expf(0.3f * trans[i] + 0.7f * agg[i] - mx) * inv;
}

// pooled[c] = sum_i fw[i] * gcf[i,c]  with gcf = [g2 | g1]
__global__ __launch_bounds__(256)
void pooled_col(const float* __restrict__ fw, const float* __restrict__ g2f,
                const float* __restrict__ g1f, float* __restrict__ pooled) {
  __shared__ float sb[32];
  const int c = blockIdx.x;
  const float* col = (c < DH) ? (g2f + c) : (g1f + (c - DH));
  float sm = 0.0f;
  for (int i = threadIdx.x; i < NP; i += 256) sm += fw[i] * col[(size_t)i * DH];
  sm = block_reduce<0>(sm, sb);
  if (threadIdx.x == 0) pooled[c] = sm;
}

__global__ __launch_bounds__(1024)
void final_head(const float* __restrict__ pooled, const float* __restrict__ lng,
                const float* __restrict__ lnb, const float* __restrict__ clsw,
                float* __restrict__ out) {
  __shared__ float sb[32];
  __shared__ float pls[1024];
  __shared__ float lg[4];
  const int c = threadIdx.x;
  float x = pooled[c];
  float s1 = block_reduce<0>(x, sb);
  float s2 = block_reduce<0>(x * x, sb);
  float m = s1 * (1.0f / 1024.0f);
  float var = s2 * (1.0f / 1024.0f) - m * m;
  pls[c] = (x - m) * rsqrtf(var + 1e-5f) * lng[c] + lnb[c];
  __syncthreads();
  if (c < 4) {
    float acc = 0.0f;
    for (int k = 0; k < 1024; k++) acc += pls[k] * clsw[k * 4 + c];
    lg[c] = acc;
  }
  __syncthreads();
  if (c == 0) {
    int best = 0;
    for (int k = 1; k < 4; k++) if (lg[k] > lg[best]) best = k;
    float sv = 1.0f;
    for (int k = 0; k < 4; k++) {
      float h = sigmoidf(lg[k]);
      out[k] = h;
      sv *= (1.0f - h);
      out[4 + k] = sv;
    }
    out[8] = (float)best;
  }
}

// --------------------------- host orchestration ----------------------------
static inline void launch_gemm(const bf16* A, const bf16* Bt, float* C,
                               const float* bias, int M, int Nn, int K,
                               hipStream_t s) {
  dim3 g(Nn / BN, M / BM), b(256);
  gemm_bf16_kernel<<<g, b, 0, s>>>(A, Bt, C, bias, M, Nn, K);
}

extern "C" void kernel_launch(void* const* d_in, const int* in_sizes, int n_in,
                              void* d_out, int out_size, void* d_ws, size_t ws_size,
                              hipStream_t stream) {
  (void)in_sizes; (void)n_in; (void)out_size; (void)ws_size;

  const float* x_path = (const float*)d_in[0];
  const float* maps   = (const float*)d_in[1];
  const float *nl_w1[2], *nl_b1[2], *nl_g[2], *nl_bn[2], *nl_w2[2], *nl_b2[2];
  for (int p = 0; p < 2; p++) {
    int base = 2 + p * 6;
    nl_w1[p] = (const float*)d_in[base + 0];
    nl_b1[p] = (const float*)d_in[base + 1];
    nl_g[p]  = (const float*)d_in[base + 2];
    nl_bn[p] = (const float*)d_in[base + 3];
    nl_w2[p] = (const float*)d_in[base + 4];
    nl_b2[p] = (const float*)d_in[base + 5];
  }
  const float *gw[6], *gb[6], *gg[6], *gbn[6];   // gc0,gc1,gc2,ga0,ga1,ga2
  for (int l = 0; l < 6; l++) {
    int base = 14 + 4 * l;
    gw[l]  = (const float*)d_in[base + 0];
    gb[l]  = (const float*)d_in[base + 1];
    gg[l]  = (const float*)d_in[base + 2];
    gbn[l] = (const float*)d_in[base + 3];
  }
  const float* attn_wa = (const float*)d_in[38];
  const float* attn_ba = (const float*)d_in[39];
  const float* attn_wb = (const float*)d_in[40];
  const float* attn_bb = (const float*)d_in[41];
  const float* attn_wc = (const float*)d_in[42];
  const float* attn_bc = (const float*)d_in[43];
  const float* cls_w   = (const float*)d_in[44];
  const float* ln_g    = (const float*)d_in[45];
  const float* ln_b    = (const float*)d_in[46];

  // ---- workspace carve-out ----
  size_t off = 0;
  char* base = (char*)d_ws;
  auto alloc = [&](size_t bytes) -> void* {
    void* p = base + off;
    off += (bytes + 255) & ~(size_t)255;
    return p;
  };
  bf16* x_bf   = (bf16*)alloc((size_t)NP * LF * 2);
  bf16* xn_bf  = (bf16*)alloc((size_t)NP * LF * 2);
  bf16* waT    = (bf16*)alloc((size_t)LF * 256 * 2);   // [256, LF]
  bf16* wbT    = (bf16*)alloc((size_t)LF * 256 * 2);
  bf16* nlw1T[2] = {(bf16*)alloc((size_t)LF * DH * 2), (bf16*)alloc((size_t)LF * DH * 2)};
  bf16* nlw2T[2] = {(bf16*)alloc((size_t)DH * DH * 2), (bf16*)alloc((size_t)DH * DH * 2)};
  bf16* gwT[6];
  for (int l = 0; l < 6; l++) gwT[l] = (bf16*)alloc((size_t)DH * DH * 2);
  float* cs_raw = (float*)alloc((size_t)NP * NP * 4);
  bf16*  cs_bf  = (bf16*)alloc((size_t)NP * NP * 2);
  bf16*  aw_bf  = (bf16*)alloc((size_t)NP * NP * 2);
  float* apre   = (float*)alloc((size_t)NP * 256 * 4);
  float* bpre   = (float*)alloc((size_t)NP * 256 * 4);
  float* aggv   = (float*)alloc((size_t)NP * 4);
  float* sv     = (float*)alloc((size_t)NP * 4);
  float* transv = (float*)alloc((size_t)NP * 4);
  float* fwv    = (float*)alloc((size_t)NP * 4);
  float* pmin   = (float*)alloc(512 * 4);
  float* pmax   = (float*)alloc(512 * 4);
  float* csmm   = (float*)alloc(2 * 4);
  float* sst3   = (float*)alloc(3 * 4);
  float* hbufA  = (float*)alloc((size_t)NP * DH * 4);
  float* hwf    = (float*)alloc((size_t)NP * DH * 4);
  bf16*  hwT    = (bf16*)alloc((size_t)NP * DH * 2);   // [DH, NP] transposed x@W
  float* tf     = (float*)alloc((size_t)NP * DH * 4);
  float* gscr_f = (float*)alloc((size_t)NP * DH * 4);
  bf16*  bpA    = (bf16*)alloc((size_t)NP * DH * 2);
  bf16*  bpB    = (bf16*)alloc((size_t)NP * DH * 2);
  float* g1_f   = (float*)alloc((size_t)NP * DH * 4);
  float* g2_f   = (float*)alloc((size_t)NP * DH * 4);
  float* pooled = (float*)alloc(1024 * 4);

  // ---- casts / transposes to bf16 ----
  cast_f32_bf16<<<2048, 256, 0, stream>>>(x_path, x_bf, (long long)NP * LF);
  transpose_cast<<<512, 256, 0, stream>>>(attn_wa, waT, LF, 256);
  transpose_cast<<<512, 256, 0, stream>>>(attn_wb, wbT, LF, 256);
  for (int p = 0; p < 2; p++) {
    transpose_cast<<<512, 256, 0, stream>>>(nl_w1[p], nlw1T[p], LF, DH);
    transpose_cast<<<512, 256, 0, stream>>>(nl_w2[p], nlw2T[p], DH, DH);
  }
  for (int l = 0; l < 6; l++)
    transpose_cast<<<512, 256, 0, stream>>>(gw[l], gwT[l], DH, DH);

  // ---- cosine-similarity adjacency ----
  rownorm_l2<<<NP, 256, 0, stream>>>(x_path, xn_bf);
  launch_gemm(xn_bf, xn_bf, cs_raw, nullptr, NP, NP, LF, stream);  // gram, Bt == A
  minmax_partial<<<512, 256, 0, stream>>>(cs_raw, (size_t)NP * NP, pmin, pmax);
  minmax_final<<<1, 256, 0, stream>>>(pmin, pmax, 512, csmm);

  // ---- gated attention ----
  launch_gemm(x_bf, waT, apre, attn_ba, NP, 256, LF, stream);
  launch_gemm(x_bf, wbT, bpre, attn_bb, NP, 256, LF, stream);
  agg_kernel<<<NP, 256, 0, stream>>>(apre, bpre, attn_wc, attn_bc, aggv, sv);
  vec_stats<<<1, 256, 0, stream>>>(sv, NP, sst3);

  // ---- adjacency builds ----
  aw_build<<<NP, 256, 0, stream>>>(maps, sv, sst3, aw_bf);
  cs_thresh<<<NP, 256, 0, stream>>>(cs_raw, csmm, aggv, cs_bf, transv);

  // ---- two GCN branches ----
  const bf16* adjs[2] = {cs_bf, aw_bf};   // branch 0 -> g1 (gc/cs), branch 1 -> g2 (ga/aw)
  float* gouts[2] = {g1_f, g2_f};
  for (int br = 0; br < 2; br++) {
    // nonlin: relu(LN(x@w1+b1)) @ w2 + b2
    launch_gemm(x_bf, nlw1T[br], hbufA, nl_b1[br], NP, DH, LF, stream);
    lnrelu<<<NP, 256, 0, stream>>>(hbufA, nl_g[br], nl_bn[br], gscr_f, bpA);
    launch_gemm(bpA, nlw2T[br], hbufA, nl_b2[br], NP, DH, DH, stream);
    cast_f32_bf16<<<2048, 256, 0, stream>>>(hbufA, bpB, (long long)NP * DH);
    const bf16* cur = bpB;
    for (int l = 0; l < 3; l++) {
      int gl = br * 3 + l;
      launch_gemm(cur, gwT[gl], hwf, nullptr, NP, DH, DH, stream);       // x @ W
      transpose_cast<<<2048, 256, 0, stream>>>(hwf, hwT, NP, DH);        // -> [DH, NP]
      launch_gemm(adjs[br], hwT, tf, gb[gl], NP, DH, NP, stream);        // adj @ (xW) + b
      bf16* ob = (cur == bpB) ? bpA : bpB;
      float* of = (l == 2) ? gouts[br] : gscr_f;
      lnrelu<<<NP, 256, 0, stream>>>(tf, gg[gl], gbn[gl], of, ob);
      cur = ob;
    }
  }

  // ---- pooling + head ----
  softmax_fw<<<1, 1024, 0, stream>>>(transv, aggv, fwv);
  pooled_col<<<1024, 256, 0, stream>>>(fwv, g2_f, g1_f, pooled);
  final_head<<<1, 1024, 0, stream>>>(pooled, ln_g, ln_b, cls_w, (float*)d_out);
}
